// GCN_3332894622188
// MI455X (gfx1250) — matrix operand order repaired and checked by reading, not compiled, not run on hardware
//
#include <hip/hip_runtime.h>
#include <hip/hip_bf16.h>

// ---------------------------------------------------------------------------
// 2-layer GCN for MI455X (gfx1250, wave32).
//   GEMMs:        v_wmma_f32_16x16x32_bf16; B panel staged in LDS per block
//                 (144B padded rows -> near-conflict-free LDS reads),
//                 A streamed from global with global_prefetch_b8.
//   Aggregation:  float4 gather + global_atomic_add_f32 scatter (L2 resident)
// ---------------------------------------------------------------------------

typedef __bf16 bf16;
typedef __attribute__((ext_vector_type(16))) __bf16 v16bf;
typedef __attribute__((ext_vector_type(8)))  __bf16 v8bf;
typedef __attribute__((ext_vector_type(8)))  float  v8f;

#define NNODES 50000
#define NEDGES 800000
#define F_IN   512
#define F_HID  256
#define F_OUT  16

// ---------------- elementwise helpers ----------------

__global__ void k_f32_to_bf16(const float* __restrict__ src,
                              bf16* __restrict__ dst, int n) {
  int i = blockIdx.x * blockDim.x + threadIdx.x;
  if (i < n) dst[i] = (bf16)src[i];
}

__global__ void k_degree(const long long* __restrict__ dst_idx,
                         float* __restrict__ deg, int E) {
  int i = blockIdx.x * blockDim.x + threadIdx.x;
  if (i < E) atomicAdd(&deg[(int)dst_idx[i]], 1.0f);
}

__global__ void k_dinv(const float* __restrict__ deg,
                       float* __restrict__ dinv, int n) {
  int i = blockIdx.x * blockDim.x + threadIdx.x;
  if (i < n) dinv[i] = rsqrtf(deg[i] + 1.0f);  // +1 self loop; always > 0
}

// ---------------- WMMA GEMM: C[M,NW] = A[M,K](bf16) x B[K,NW](bf16) ----------
// Block = 8 waves = 128 M rows; all waves share one BN=NTILES*16 column panel.
// B panel staged cooperatively into LDS (row stride BN+8 elems = +16B pad so
// per-lane fragment reads spread across banks), then the K loop feeds WMMA
// from per-element bf16 LDS reads. A fragments stream from global (per-lane rows) with
// prefetch. WMMA layouts per CDNA5 ISA (wave32):
//   A: lane m<16 holds row m; elems 0..7 -> K=kk+8*half.., elems 8..15 -> +16
//   B: lane holds col N=lane&15; elems 0..7 -> K=kk+8*half.., elems 8..15 -> +16
//   C: vgpr v -> M = v + 8*half, N = lane&15

template<int K, int NW, int NTILES>
__global__ void __launch_bounds__(256)
k_wmma_gemm(const bf16* __restrict__ A, const bf16* __restrict__ B,
            float* __restrict__ C, int M) {
  constexpr int BN     = NTILES * 16;   // panel width (cols)
  constexpr int LDSROW = BN + 8;        // padded row stride (elements)
  extern __shared__ char smem_raw[];
  bf16* bs = (bf16*)smem_raw;           // [K][LDSROW]

  const int lane = threadIdx.x & 31;
  const int wave = threadIdx.x >> 5;
  const int half = lane >> 4;
  const int ml   = lane & 15;
  const int m0   = (blockIdx.x * (blockDim.x >> 5) + wave) * 16;
  const int n0   = blockIdx.y * BN;

  // ---- cooperative stage: global B panel -> LDS (16B chunks) ----
  constexpr int CPR = BN / 8;           // 16B chunks per row
  for (int c = threadIdx.x; c < K * CPR; c += blockDim.x) {
    int k  = c / CPR;
    int cc = c % CPR;
    *(uint4*)(bs + (size_t)k * LDSROW + cc * 8) =
        *(const uint4*)(B + (size_t)k * NW + n0 + cc * 8);
  }
  __syncthreads();
  if (m0 >= M) return;                  // wave-uniform; no barriers after this

  v8f acc[NTILES];
#pragma unroll
  for (int t = 0; t < NTILES; ++t)
    acc[t] = v8f{0.f, 0.f, 0.f, 0.f, 0.f, 0.f, 0.f, 0.f};

  const bf16* __restrict__ arow = A + (size_t)(m0 + ml) * K + half * 8;

#pragma unroll
  for (int kk = 0; kk < K; kk += 32) {
    if (kk + 64 <= K)
      __builtin_prefetch(arow + kk + 64, 0, 0);   // global_prefetch_b8

    v8bf alo = *(const v8bf*)(arow + kk);
    v8bf ahi = *(const v8bf*)(arow + kk + 16);
    v16bf af = __builtin_shufflevector(alo, ahi, 0, 1, 2, 3, 4, 5, 6, 7,
                                       8, 9, 10, 11, 12, 13, 14, 15);

    const bf16* brow = bs + (size_t)(kk + half * 8) * LDSROW + ml;
#pragma unroll
    for (int t = 0; t < NTILES; ++t) {
      v16bf bf;
#pragma unroll
      for (int s = 0; s < 16; ++s)
        bf[s] = brow[(size_t)(s < 8 ? s : s + 8) * LDSROW + t * 16];
      acc[t] = __builtin_amdgcn_wmma_f32_16x16x32_bf16(
          false, af, false, bf, (short)0, acc[t], false, false);
    }
  }

#pragma unroll
  for (int t = 0; t < NTILES; ++t) {
    float* crow = C + (size_t)m0 * NW + n0 + t * 16 + ml;
#pragma unroll
    for (int v = 0; v < 8; ++v)
      crow[(size_t)(v + 8 * half) * NW] = acc[t][v];
  }
}

// ---------------- edge scatter: agg[dst] += h[src] * dinv[src]*dinv[dst] ----
// float4 per thread: b128 gather, 4 x global_atomic_add_f32 scatter.

template<int F>
__global__ void k_edge_agg(const float* __restrict__ h,
                           const float* __restrict__ dinv,
                           const long long* __restrict__ src,
                           const long long* __restrict__ dst,
                           float* __restrict__ agg, int E) {
  const int TPE = F / 4;                    // threads per edge
  long long idx = (long long)blockIdx.x * blockDim.x + threadIdx.x;
  if (idx >= (long long)E * TPE) return;
  int e  = (int)(idx / TPE);
  int f4 = (int)(idx % TPE) * 4;
  int s = (int)src[e];
  int d = (int)dst[e];
  float w = dinv[s] * dinv[d];
  float4 v = *(const float4*)(h + (size_t)s * F + f4);
  float* p = agg + (size_t)d * F + f4;
  atomicAdd(p + 0, v.x * w);
  atomicAdd(p + 1, v.y * w);
  atomicAdd(p + 2, v.z * w);
  atomicAdd(p + 3, v.w * w);
}

// layer-1 epilogue: + self-loop + bias, ReLU, convert to bf16 for GEMM2
__global__ void k_finish1(const float* __restrict__ agg,
                          const float* __restrict__ h1,
                          const float* __restrict__ dinv,
                          const float* __restrict__ b1,
                          bf16* __restrict__ hb, int total) {
  int i = blockIdx.x * blockDim.x + threadIdx.x;
  if (i >= total) return;
  int n = i >> 8;            // F_HID == 256
  int f = i & 255;
  float di = dinv[n];
  float v = agg[i] + h1[i] * di * di + b1[f];
  hb[i] = (bf16)fmaxf(v, 0.0f);
}

// layer-2 epilogue: + self-loop + bias -> fp32 output
__global__ void k_finish2(const float* __restrict__ agg,
                          const float* __restrict__ p2,
                          const float* __restrict__ dinv,
                          const float* __restrict__ b2,
                          float* __restrict__ out, int total) {
  int i = blockIdx.x * blockDim.x + threadIdx.x;
  if (i >= total) return;
  int n = i >> 4;            // F_OUT == 16
  int f = i & 15;
  float di = dinv[n];
  out[i] = agg[i] + p2[i] * di * di + b2[f];
}

// ---------------------------------------------------------------------------

extern "C" void kernel_launch(void* const* d_in, const int* in_sizes, int n_in,
                              void* d_out, int out_size, void* d_ws, size_t ws_size,
                              hipStream_t stream) {
  (void)in_sizes; (void)n_in; (void)out_size; (void)ws_size;

  const float*     x   = (const float*)d_in[0];                 // [N, 512]
  const long long* ei  = (const long long*)d_in[1];             // [2, E] int64
  const float*     W1  = (const float*)d_in[2];                 // [512, 256]
  const float*     b1  = (const float*)d_in[3];                 // [256]
  const float*     W2  = (const float*)d_in[4];                 // [256, 16]
  const float*     b2  = (const float*)d_in[5];                 // [16]
  const long long* srcI = ei;
  const long long* dstI = ei + NEDGES;
  float* out = (float*)d_out;                                   // [N, 16]

  // ---- workspace bump allocator (1KB aligned) ----
  char* ws = (char*)d_ws;
  size_t off = 0;
  auto alloc = [&](size_t bytes) -> char* {
    char* p = ws + off;
    off = (off + bytes + 1023) & ~(size_t)1023;
    return p;
  };
  float* deg  = (float*)alloc((size_t)NNODES * 4);
  float* dinv = (float*)alloc((size_t)NNODES * 4);
  bf16*  w1b  = (bf16*) alloc((size_t)F_IN * F_HID * 2);
  bf16*  w2b  = (bf16*) alloc((size_t)F_HID * F_OUT * 2);
  char*  xb_raw = alloc((size_t)NNODES * F_IN * 2);   // 51.2 MB
  bf16*  xb   = (bf16*)xb_raw;
  float* agg1 = (float*)xb_raw;                       // alias: dead after GEMM1
  float* h1   = (float*)alloc((size_t)NNODES * F_HID * 4);
  bf16*  hb   = (bf16*) alloc((size_t)NNODES * F_HID * 2);
  float* p2   = (float*)alloc((size_t)NNODES * F_OUT * 4);
  float* agg2 = (float*)alloc((size_t)NNODES * F_OUT * 4);

  const int B = 256;

  // degree + dinv
  hipMemsetAsync(deg, 0, (size_t)NNODES * 4, stream);
  k_degree<<<(NEDGES + B - 1) / B, B, 0, stream>>>(dstI, deg, NEDGES);
  k_dinv  <<<(NNODES + B - 1) / B, B, 0, stream>>>(deg, dinv, NNODES);

  // fp32 -> bf16 operands
  k_f32_to_bf16<<<(NNODES * F_IN + B - 1) / B, B, 0, stream>>>(x, xb, NNODES * F_IN);
  k_f32_to_bf16<<<(F_IN * F_HID + B - 1) / B, B, 0, stream>>>(W1, w1b, F_IN * F_HID);
  k_f32_to_bf16<<<(F_HID * F_OUT + B - 1) / B, B, 0, stream>>>(W2, w2b, F_HID * F_OUT);

  // GEMM1: h1[N,256] = xb @ w1b   (8 waves x 16 rows; 64-col panel; grid.y = 4)
  {
    const int mtiles = NNODES / 16;                 // 3125
    dim3 grid((mtiles + 7) / 8, F_HID / 64);
    constexpr size_t lds1 = (size_t)F_IN * (4 * 16 + 8) * sizeof(bf16); // 73728
    k_wmma_gemm<F_IN, F_HID, 4><<<grid, 256, lds1, stream>>>(xb, w1b, h1, NNODES);
  }

  // edge aggregation, layer 1 (agg1 aliases xb; memset ordered after GEMM1)
  hipMemsetAsync(agg1, 0, (size_t)NNODES * F_HID * 4, stream);
  {
    long long tot = (long long)NEDGES * (F_HID / 4);
    k_edge_agg<F_HID><<<(unsigned)((tot + B - 1) / B), B, 0, stream>>>(
        h1, dinv, srcI, dstI, agg1, NEDGES);
  }
  k_finish1<<<(NNODES * F_HID + B - 1) / B, B, 0, stream>>>(
      agg1, h1, dinv, b1, hb, NNODES * F_HID);

  // GEMM2: p2[N,16] = hb @ w2b   (single 16-wide N tile)
  {
    const int mtiles = NNODES / 16;
    dim3 grid((mtiles + 7) / 8, 1);
    constexpr size_t lds2 = (size_t)F_HID * (1 * 16 + 8) * sizeof(bf16); // 12288
    k_wmma_gemm<F_HID, F_OUT, 1><<<grid, 256, lds2, stream>>>(hb, w2b, p2, NNODES);
  }

  // edge aggregation, layer 2
  hipMemsetAsync(agg2, 0, (size_t)NNODES * F_OUT * 4, stream);
  {
    long long tot = (long long)NEDGES * (F_OUT / 4);
    k_edge_agg<F_OUT><<<(unsigned)((tot + B - 1) / B), B, 0, stream>>>(
        p2, dinv, srcI, dstI, agg2, NEDGES);
  }
  k_finish2<<<(NNODES * F_OUT + B - 1) / B, B, 0, stream>>>(
      agg2, p2, dinv, b2, out, NNODES * F_OUT);
}
